// AttentionBlock_9947144258160
// MI455X (gfx1250) — matrix operand order repaired and checked by reading, not compiled, not run on hardware
//
#include <hip/hip_runtime.h>
#include <hip/hip_bf16.h>

typedef _Float16 f16;
typedef __attribute__((ext_vector_type(16))) _Float16 v16h;
typedef __attribute__((ext_vector_type(8)))  _Float16 v8h;
typedef __attribute__((ext_vector_type(8)))  float    v8f;
typedef __attribute__((ext_vector_type(4)))  int      v4i;

#define DEVINL static __device__ __forceinline__

constexpr int kB   = 16;    // batch
constexpr int kC   = 512;   // channels
constexpr int kH   = 8;     // heads
constexpr int kCH  = 64;    // channels per head
constexpr int kT   = 1024;  // tokens (32*32)
constexpr int kTilePitch = 40;   // f16 pitch of 64x32 LDS tiles (bank stride 20 dwords)
constexpr int kLogitLd   = 1032; // f32 pitch of logits rows (breaks bank aliasing)
constexpr float kAttScale = 0.35355339059327373f; // 1/sqrt(sqrt(64))

// ---------------------------------------------------------------------------
// CDNA5 async Global->LDS copy (ASYNCcnt) with toolchain-portable fallback
// ---------------------------------------------------------------------------
DEVINL void async_b128(const f16* gp, f16* lp) {
#if __has_builtin(__builtin_amdgcn_global_load_async_to_lds_b128)
  __builtin_amdgcn_global_load_async_to_lds_b128(
      (__attribute__((address_space(1))) v4i*)gp,
      (__attribute__((address_space(3))) v4i*)lp, 0, 0);
#else
  asm volatile("global_load_async_to_lds_b128 %0, %1, off"
               :: "v"((unsigned)(size_t)(__attribute__((address_space(3))) void*)lp),
                  "v"(gp)
               : "memory");
#endif
}

DEVINL void wait_async0() {
#if __has_builtin(__builtin_amdgcn_s_wait_asynccnt)
  __builtin_amdgcn_s_wait_asynccnt(0);
#else
  asm volatile("s_wait_asynccnt 0x0" ::: "memory");
#endif
}

// Copy a 64-row x 32-f16 tile (row stride ldg) into an LDS tile with
// kTilePitch row pitch. 128 threads x 2 x 16B async transfers.
DEVINL void async_tile_load(const f16* g0, int ldg, f16* l0) {
  const int l    = threadIdx.x;   // 0..127
  const int part = l & 3;         // 16B chunk within 64B row
  const int r0   = l >> 2;        // 0..31
#pragma unroll
  for (int rr = 0; rr < 2; ++rr) {
    const int row = r0 + rr * 32;
    async_b128(g0 + (size_t)row * ldg + part * 8,
               l0 + row * kTilePitch + part * 8);
  }
}

// ---------------------------------------------------------------------------
// WMMA fragment helpers (layouts per CDNA5 ISA §7.12.2, wave32)
// ---------------------------------------------------------------------------

// A-matrix 16x32 f16: lane m = lane&15, kb = (lane>=16)?8:0;
// VGPR0..3 <- K kb..kb+7, VGPR4..7 <- K kb+16..kb+23.
DEVINL v16h load_a_row(const f16* base, int lda) {
  const int lane = threadIdx.x & 31;
  const int m  = lane & 15;
  const int kb = (lane >> 4) << 3;
  const f16* p = base + (size_t)m * lda + kb;
  v8h lo = *(const v8h*)(p);
  v8h hi = *(const v8h*)(p + 16);
  v16h r;
#pragma unroll
  for (int i = 0; i < 8; ++i) { r[i] = lo[i]; r[i + 8] = hi[i]; }
  return r;
}

// B-matrix 32x16 f16 (KxN) from N-major memory: element (k,n) at base[n*ldb+k].
// lane n = lane&15, kb = (lane>=16)?8:0; VGPR0..3 <- K kb..kb+7, VGPR4..7 <- K kb+16..kb+23.
DEVINL v16h load_b_nmajor(const f16* base, int ldb) {
  const int lane = threadIdx.x & 31;
  const int n  = lane & 15;
  const int kb = (lane >> 4) << 3;
  const f16* p = base + (size_t)n * ldb + kb;
  v8h lo = *(const v8h*)(p);
  v8h hi = *(const v8h*)(p + 16);
  v16h r;
#pragma unroll
  for (int i = 0; i < 8; ++i) { r[i] = lo[i]; r[i + 8] = hi[i]; }
  return r;
}

DEVINL v8f wmma_f16(v16h a, v16h b, v8f c) {
  return __builtin_amdgcn_wmma_f32_16x16x32_f16(
      /*neg_a=*/false, a, /*neg_b=*/false, b,
      /*c_mod=*/(short)0, c, /*reuse_a=*/false, /*reuse_b=*/false);
}

DEVINL v8f vzero8() { v8f z = {0.f,0.f,0.f,0.f,0.f,0.f,0.f,0.f}; return z; }

// ---------------------------------------------------------------------------
// Kernel 1: fp32 -> f16 weight conversion
// ---------------------------------------------------------------------------
__global__ __launch_bounds__(256)
void cvt_weights_kernel(const float* __restrict__ wq, const float* __restrict__ wp,
                        f16* __restrict__ wq16, f16* __restrict__ wp16) {
  const int i = blockIdx.x * 256 + threadIdx.x;
  const int nq = 3 * kC * kC;
  const int np = kC * kC;
  if (i < nq)            wq16[i] = (f16)wq[i];
  else if (i < nq + np)  wp16[i - nq] = (f16)wp[i - nq];
}

// ---------------------------------------------------------------------------
// Kernel 2: GroupNorm; writes xn^T f16 as [b][t][C] (N-major B operand).
// ---------------------------------------------------------------------------
__global__ __launch_bounds__(256)
void groupnorm_kernel(const float* __restrict__ x,
                      const float* __restrict__ sc, const float* __restrict__ bi,
                      f16* __restrict__ xnT) {
  const int g = blockIdx.x;
  const int b = blockIdx.y;
  const int tid = threadIdx.x;
  const float* xb = x + ((size_t)b * kC + (size_t)g * 16) * kT;

  float s1 = 0.f, s2 = 0.f;
  for (int i = tid; i < 16 * kT; i += 256) {
    float v = xb[i]; s1 += v; s2 += v * v;
  }
  __shared__ float r1[256], r2[256];
  r1[tid] = s1; r2[tid] = s2; __syncthreads();
  for (int st = 128; st > 0; st >>= 1) {
    if (tid < st) { r1[tid] += r1[tid + st]; r2[tid] += r2[tid + st]; }
    __syncthreads();
  }
  const float inv_n = 1.f / (16.f * kT);
  const float mu   = r1[0] * inv_n;
  const float var  = r2[0] * inv_n - mu * mu;
  const float rstd = rsqrtf(var + 1e-5f);

  for (int i = tid; i < 16 * kT; i += 256) {
    const int cl = i >> 10;
    const int tt = i & (kT - 1);
    const int c  = g * 16 + cl;
    const float v = (xb[i] - mu) * rstd * sc[c] + bi[c];
    xnT[((size_t)b * kT + tt) * kC + c] = (f16)v;
  }
}

// ---------------------------------------------------------------------------
// Shared GEMM core: 64x64 block tile, K=512 in 16 steps of 32.
// A and B 64x32 tiles staged in LDS via async copies, double-buffered so the
// prefetch of step ks+1 overlaps the WMMAs of step ks. 4 waves x 16 M rows,
// each wave covers 4 N subtiles.
// ---------------------------------------------------------------------------
DEVINL void gemm_core_64x64(const f16* gA, const f16* gB,
                            f16 (*sA)[64 * kTilePitch], f16 (*sB)[64 * kTilePitch],
                            int wave, v8f acc[4]) {
  async_tile_load(gA, kC, sA[0]);
  async_tile_load(gB, kC, sB[0]);
  wait_async0();
  __syncthreads();
#pragma unroll 2
  for (int ks = 0; ks < kC / 32; ++ks) {
    const int cur = ks & 1;
    if (ks + 1 < kC / 32) {
      async_tile_load(gA + (ks + 1) * 32, kC, sA[cur ^ 1]);
      async_tile_load(gB + (ks + 1) * 32, kC, sB[cur ^ 1]);
    }
    v16h a = load_a_row(sA[cur] + (wave * 16) * kTilePitch, kTilePitch);
#pragma unroll
    for (int j = 0; j < 4; ++j) {
      v16h bf = load_b_nmajor(sB[cur] + (j * 16) * kTilePitch, kTilePitch);
      acc[j] = wmma_f16(a, bf, acc[j]);
    }
    wait_async0();       // prefetch landed
    __syncthreads();     // everyone done reading cur
  }
}

// ---------------------------------------------------------------------------
// Kernel 3: QKV GEMM. q,k -> transposed [bh][t][64] with sqrt-scale folded;
// v -> channel-major [bh][64][t].
// ---------------------------------------------------------------------------
__global__ __launch_bounds__(128)
void qkv_gemm_kernel(const f16* __restrict__ wq16, const float* __restrict__ bqkv,
                     const f16* __restrict__ xnT,
                     f16* __restrict__ qT, f16* __restrict__ kT_, f16* __restrict__ vO) {
  __shared__ __align__(16) f16 sA[2][64 * kTilePitch];
  __shared__ __align__(16) f16 sB[2][64 * kTilePitch];

  const int nb = blockIdx.x;             // t block (64 wide)
  const int mb = blockIdx.y;             // o block (64 tall)
  const int b  = blockIdx.z;
  const int wave = threadIdx.x >> 5;
  const int lane = threadIdx.x & 31;
  const int o0 = mb * 64 + wave * 16;

  const f16* gA = wq16 + (size_t)(mb * 64) * kC;
  const f16* gB = xnT + ((size_t)b * kT + (size_t)nb * 64) * kC;

  v8f acc[4] = {vzero8(), vzero8(), vzero8(), vzero8()};
  gemm_core_64x64(gA, gB, sA, sB, wave, acc);

  const int n  = lane & 15;
  const int mr = (lane >> 4) << 3;
  float bias8[8];
#pragma unroll
  for (int r = 0; r < 8; ++r) bias8[r] = bqkv[o0 + mr + r];

  const int seg  = o0 >> 9;              // 0=q, 1=k, 2=v
  const int ol   = o0 & (kC - 1);
  const int head = ol >> 6;
  const int cc0  = ol & (kCH - 1);
  const size_t bh = (size_t)b * kH + head;

#pragma unroll
  for (int j = 0; j < 4; ++j) {
    const int tg = nb * 64 + j * 16 + n;
    if (seg < 2) {
      f16* dstT = (seg == 0 ? qT : kT_) + bh * kT * kCH;
      v8h p;
#pragma unroll
      for (int r = 0; r < 8; ++r)
        p[r] = (f16)((acc[j][r] + bias8[r]) * kAttScale);
      *(v8h*)(dstT + (size_t)tg * kCH + cc0 + mr) = p;
    } else {
      f16* dv = vO + bh * (size_t)kCH * kT;
#pragma unroll
      for (int r = 0; r < 8; ++r)
        dv[(size_t)(cc0 + mr + r) * kT + tg] = (f16)(acc[j][r] + bias8[r]);
    }
  }
}

// ---------------------------------------------------------------------------
// Kernel 4: fused attention per (bh, 16-row t-block).
// ---------------------------------------------------------------------------
__global__ __launch_bounds__(128)
void attn_kernel(const f16* __restrict__ qT, const f16* __restrict__ kT_,
                 const f16* __restrict__ vO, f16* __restrict__ aT) {
  extern __shared__ float smem[];
  float* logits  = smem;                     // 16 * kLogitLd
  float* red     = smem + 16 * kLogitLd;     // 16*8 partials
  float* rowmax  = red + 128;                // 16
  float* rowrinv = rowmax + 16;              // 16

  const int tb = blockIdx.x;
  const int bh = blockIdx.y;
  const int tid  = threadIdx.x;
  const int wave = tid >> 5;
  const int lane = tid & 31;

  const f16* qh = qT  + (size_t)bh * kT * kCH + (size_t)tb * 16 * kCH;
  const f16* kh = kT_ + (size_t)bh * kT * kCH;
  const f16* vh = vO  + (size_t)bh * (size_t)kCH * kT;

  // ---- Phase 1: logits = q^T k ----
  const v16h a0 = load_a_row(qh + 0,  kCH);
  const v16h a1 = load_a_row(qh + 32, kCH);
  const int n  = lane & 15;
  const int mr = (lane >> 4) << 3;
  for (int i = 0; i < 16; ++i) {
    const int st = wave * 16 + i;
    v8f acc = vzero8();
    v16h b0 = load_b_nmajor(kh + (size_t)st * 16 * kCH + 0,  kCH);
    v16h b1 = load_b_nmajor(kh + (size_t)st * 16 * kCH + 32, kCH);
    acc = wmma_f16(a0, b0, acc);
    acc = wmma_f16(a1, b1, acc);
#pragma unroll
    for (int r = 0; r < 8; ++r)
      logits[(size_t)(mr + r) * kLogitLd + st * 16 + n] = acc[r];
  }
  __syncthreads();

  // ---- Phase 2: softmax (8 threads per row) ----
  const int row = tid >> 3;
  const int sub = tid & 7;
  float mx = -3.4e38f;
  for (int i = 0; i < kT / 8; ++i)
    mx = fmaxf(mx, logits[(size_t)row * kLogitLd + sub + 8 * i]);
  red[row * 8 + sub] = mx;
  __syncthreads();
  if (sub == 0) {
    float m = red[row * 8];
#pragma unroll
    for (int j = 1; j < 8; ++j) m = fmaxf(m, red[row * 8 + j]);
    rowmax[row] = m;
  }
  __syncthreads();
  const float rm = rowmax[row];
  float sum = 0.f;
  for (int i = 0; i < kT / 8; ++i) {
    const size_t idx = (size_t)row * kLogitLd + sub + 8 * i;
    const float e = __expf(logits[idx] - rm);
    logits[idx] = e;
    sum += e;
  }
  red[row * 8 + sub] = sum;
  __syncthreads();
  if (sub == 0) {
    float s = 0.f;
#pragma unroll
    for (int j = 0; j < 8; ++j) s += red[row * 8 + j];
    rowrinv[row] = 1.f / s;
  }
  __syncthreads();

  // ---- Phase 3: aT[t, c] = softmax(W) . v^T ----
  {
    const int m  = lane & 15;
    const int kb = (lane >> 4) << 3;
    const float rinv = rowrinv[m];
    v8f acc = vzero8();
    for (int ks = 0; ks < kT / 32; ++ks) {
      v16h a;
      const float* lp = logits + (size_t)m * kLogitLd + ks * 32 + kb;
#pragma unroll
      for (int i = 0; i < 8; ++i) {
        a[i]     = (f16)(lp[i]      * rinv);
        a[i + 8] = (f16)(lp[i + 16] * rinv);
      }
      v16h bf = load_b_nmajor(vh + (size_t)(wave * 16) * kT + ks * 32, kT);
      acc = wmma_f16(a, bf, acc);
    }
    const int b    = bh >> 3;
    const int head = bh & 7;
    f16* ao = aT + (size_t)b * kT * kC;
    const int c = head * kCH + wave * 16 + n;
#pragma unroll
    for (int r = 0; r < 8; ++r)
      ao[(size_t)(tb * 16 + mr + r) * kC + c] = (f16)acc[r];
  }
}

// ---------------------------------------------------------------------------
// Kernel 5: proj GEMM + bias + residual -> f32 output.
// ---------------------------------------------------------------------------
__global__ __launch_bounds__(128)
void proj_kernel(const f16* __restrict__ wp16, const float* __restrict__ bproj,
                 const f16* __restrict__ aT, const float* __restrict__ x,
                 float* __restrict__ out) {
  __shared__ __align__(16) f16 sA[2][64 * kTilePitch];
  __shared__ __align__(16) f16 sB[2][64 * kTilePitch];

  const int nb = blockIdx.x;
  const int mb = blockIdx.y;
  const int b  = blockIdx.z;
  const int wave = threadIdx.x >> 5;
  const int lane = threadIdx.x & 31;
  const int o0 = mb * 64 + wave * 16;

  const f16* gA = wp16 + (size_t)(mb * 64) * kC;
  const f16* gB = aT + ((size_t)b * kT + (size_t)nb * 64) * kC;

  v8f acc[4] = {vzero8(), vzero8(), vzero8(), vzero8()};
  gemm_core_64x64(gA, gB, sA, sB, wave, acc);

  const int n  = lane & 15;
  const int mr = (lane >> 4) << 3;
  float bias8[8];
#pragma unroll
  for (int r = 0; r < 8; ++r) bias8[r] = bproj[o0 + mr + r];

#pragma unroll
  for (int j = 0; j < 4; ++j) {
    const int tg = nb * 64 + j * 16 + n;
#pragma unroll
    for (int r = 0; r < 8; ++r) {
      const int o = o0 + mr + r;
      const size_t idx = ((size_t)b * kC + o) * kT + tg;
      out[idx] = x[idx] + bias8[r] + acc[j][r];
    }
  }
}

// ---------------------------------------------------------------------------
// Host launcher
// ---------------------------------------------------------------------------
extern "C" void kernel_launch(void* const* d_in, const int* in_sizes, int n_in,
                              void* d_out, int out_size, void* d_ws, size_t ws_size,
                              hipStream_t stream) {
  const float* x        = (const float*)d_in[0];
  const float* gn_scale = (const float*)d_in[1];
  const float* gn_bias  = (const float*)d_in[2];
  const float* w_qkv    = (const float*)d_in[3];
  const float* b_qkv    = (const float*)d_in[4];
  const float* w_proj   = (const float*)d_in[5];
  const float* b_proj   = (const float*)d_in[6];
  float* out = (float*)d_out;

  char* w = (char*)d_ws;
  constexpr size_t kXnTBytes = (size_t)kB * kT * kC * 2;
  constexpr size_t kWqBytes  = (size_t)3 * kC * kC * 2;
  constexpr size_t kWpBytes  = (size_t)kC * kC * 2;
  constexpr size_t kQKVBytes = (size_t)kB * kH * kT * kCH * 2;
  size_t off = 0;
  f16* xnT  = (f16*)(w + off); off += kXnTBytes;
  f16* wq16 = (f16*)(w + off); off += kWqBytes;
  f16* wp16 = (f16*)(w + off); off += kWpBytes;
  f16* qT   = (f16*)(w + off); off += kQKVBytes;
  f16* kT_  = (f16*)(w + off); off += kQKVBytes;
  f16* vO   = (f16*)(w + off); off += kQKVBytes;
  f16* aT   = (f16*)(w + off); off += kQKVBytes;
  (void)ws_size; (void)in_sizes; (void)n_in; (void)out_size;

  {
    const int total = 3 * kC * kC + kC * kC;
    cvt_weights_kernel<<<(total + 255) / 256, 256, 0, stream>>>(w_qkv, w_proj, wq16, wp16);
  }
  groupnorm_kernel<<<dim3(32, kB), 256, 0, stream>>>(x, gn_scale, gn_bias, xnT);
  qkv_gemm_kernel<<<dim3(kT / 64, (3 * kC) / 64, kB), 128, 0, stream>>>(
      wq16, b_qkv, xnT, qT, kT_, vO);
  {
    const size_t smem = (size_t)(16 * kLogitLd + 128 + 32) * sizeof(float);
    attn_kernel<<<dim3(kT / 16, kB * kH), 128, smem, stream>>>(qT, kT_, vO, aT);
  }
  proj_kernel<<<dim3(kT / 64, kC / 64, kB), 128, 0, stream>>>(wp16, b_proj, aT, x, out);
}
